// EdgeConv_42159398978122
// MI455X (gfx1250) — matrix operand order repaired and checked by reading, not compiled, not run on hardware
//
#include <hip/hip_runtime.h>
#include <hip/hip_bf16.h>

// ---------------- problem constants (from reference setup_inputs) ----------
#define BB     4
#define NN     8192
#define CFEAT  128
#define CAUG   131          // feat(128) + xyz(3)
#define SS     2048
#define KNN_K  20
#define COUT   256
#define KPAD   160          // 131 padded to 5 * 32
#define NKT    5            // KPAD / 32
#define BN_EPS 1e-5f

// ---------------- workspace layout (bytes) ---------------------------------
#define OFF_SIDX  0u          // int  [BB*SS]            = 32768 B
#define OFF_KNN   32768u      // int  [BB*SS*KNN_K]      = 655360 B
#define OFF_WTOP  688128u     // f16  [NKT*COUT*32]      = 81920 B (swizzled)
#define OFF_W2    770048u     // f16  [NKT*COUT*32]      = 81920 B (swizzled)
#define OFF_CT    851968u     // f32  [BB*SS*COUT]       = 8388608 B
#define OFF_SUM   9240576u    // f32  [COUT]
#define OFF_SQ    9241600u    // f32  [COUT]
#define OFF_SA    9242624u    // f32  [COUT]  scale = rstd*gamma
#define OFF_SB    9243648u    // f32  [COUT]  shift = beta - mean*scale

// ---------------- output layout (floats) -----------------------------------
#define OUT_XYZ   0                       // BB*SS*3   = 24576
#define OUT_FEAT  24576                   // BB*SS*256 = 2097152
#define OUT_SIDX  (24576 + 2097152)       // BB*SS     = 8192

typedef _Float16 v16h __attribute__((ext_vector_type(16)));
typedef _Float16 v8h  __attribute__((ext_vector_type(8)));
typedef float    v8f  __attribute__((ext_vector_type(8)));

// ---------------- WMMA helpers ---------------------------------------------
__device__ __forceinline__ v8f wmma_f16(v16h a, v16h b, v8f c) {
  return __builtin_amdgcn_wmma_f32_16x16x32_f16(false, a, false, b,
                                                (short)0, c, false, false);
}

// A fragment from row-major LDS tile [rows][KPAD] (f16).
// lanes 0-15: row L, elems = K[0..15]; lanes 16-31: K[16..31]
__device__ __forceinline__ v16h load_afrag(const _Float16* As, int rowBase,
                                           int kt, int lane) {
  int row = rowBase + (lane & 15);
  int o   = row * KPAD + kt * 32 + ((lane & 16) ? 16 : 0);
  v8h lo = *(const v8h*)(As + o);
  v8h hi = *(const v8h*)(As + o + 8);
  return __builtin_shufflevector(lo, hi, 0,1,2,3,4,5,6,7,8,9,10,11,12,13,14,15);
}

// B fragment from pre-swizzled global weights: Wf[kt][col][32 halfs],
// halfs 0..15 = K[kt*32+0..15] (lanes 0-15), 16..31 = K[kt*32+16..31] (lanes 16-31)
__device__ __forceinline__ v16h load_bfrag(const _Float16* Wf, int kt, int ct,
                                           int lane) {
  int col = ct * 16 + (lane & 15);
  int off = ((kt * COUT + col) << 5) + ((lane & 16) ? 16 : 0);
  return *(const v16h*)(Wf + off);
}

// ---------------- Kernel 1: furthest point sampling -------------------------
__global__ void fps_kernel(const float* __restrict__ xyz,
                           const int* __restrict__ initf,
                           int* __restrict__ sidx) {
  const int b = blockIdx.x, tid = threadIdx.x;          // 1024 threads
  __shared__ float dist[NN];
  __shared__ float wv[32];
  __shared__ int   wi[32];
  __shared__ int   curFar;
  for (int i = tid; i < NN; i += 1024) dist[i] = 1e10f;
  if (tid == 0) curFar = initf[b];
  __syncthreads();
  const float* bx = xyz + (size_t)b * NN * 3;
  for (int it = 0; it < SS; ++it) {
    int far = curFar;
    if (tid == 0) sidx[b * SS + it] = far;
    float cx = bx[far * 3 + 0], cy = bx[far * 3 + 1], cz = bx[far * 3 + 2];
    float bv = -1.0f; int bi = 0;
    for (int i = tid; i < NN; i += 1024) {
      float dx = bx[i * 3 + 0] - cx;
      float dy = bx[i * 3 + 1] - cy;
      float dz = bx[i * 3 + 2] - cz;
      float d  = dx * dx + dy * dy + dz * dz;
      float nd = fminf(dist[i], d);
      dist[i] = nd;
      if (nd > bv) { bv = nd; bi = i; }   // ascending i => first-max kept
    }
    for (int m = 16; m >= 1; m >>= 1) {
      float ov = __shfl_xor(bv, m, 32);
      int   oi = __shfl_xor(bi, m, 32);
      if (ov > bv || (ov == bv && oi < bi)) { bv = ov; bi = oi; }
    }
    if ((tid & 31) == 0) { wv[tid >> 5] = bv; wi[tid >> 5] = bi; }
    __syncthreads();
    if (tid < 32) {
      float v = wv[tid]; int ix = wi[tid];
      for (int m = 16; m >= 1; m >>= 1) {
        float ov = __shfl_xor(v, m, 32);
        int   oi = __shfl_xor(ix, m, 32);
        if (ov > v || (ov == v && oi < ix)) { v = ov; ix = oi; }
      }
      if (tid == 0) curFar = ix;
    }
    __syncthreads();
  }
}

// ---------------- Kernel 2: new_xyz + sample_idx outputs --------------------
__global__ void out_misc_kernel(const float* __restrict__ xyz,
                                const int* __restrict__ sidx,
                                float* __restrict__ out) {
  int g = blockIdx.x * 256 + threadIdx.x;
  if (g >= BB * SS) return;
  int b = g >> 11;
  int p = sidx[g];
  const float* src = xyz + ((size_t)b * NN + p) * 3;
  out[OUT_XYZ + g * 3 + 0] = src[0];
  out[OUT_XYZ + g * 3 + 1] = src[1];
  out[OUT_XYZ + g * 3 + 2] = src[2];
  out[OUT_SIDX + g] = (float)p;
}

// ---------------- Kernel 3: weight swizzle into WMMA-B layout ---------------
__global__ void prep_w_kernel(const float* __restrict__ conv_w,   // [262][256]
                              _Float16* __restrict__ wtop,
                              _Float16* __restrict__ w2) {
  int idx = blockIdx.x * 256 + threadIdx.x;               // NKT*COUT*32 entries
  if (idx >= NKT * COUT * 32) return;
  int kt  = idx / (COUT * 32);
  int rem = idx - kt * COUT * 32;
  int col = rem >> 5;
  int j   = rem & 31;
  int k   = kt * 32 + j;
  float t = 0.f, d = 0.f;
  if (k < CAUG) {
    float top = conv_w[(size_t)k * COUT + col];
    float bot = conv_w[(size_t)(CAUG + k) * COUT + col];
    t = top; d = bot - top;
  }
  wtop[idx] = (_Float16)t;
  w2[idx]   = (_Float16)d;
}

// ---------------- Kernel 4: KNN (top-20 smallest d2, stable ties) -----------
#define KTS 512
__global__ void knn_kernel(const float* __restrict__ xyz,
                           const int* __restrict__ sidx,
                           int* __restrict__ knn) {
  int g = blockIdx.x * 256 + threadIdx.x;                 // 0 .. BB*SS-1
  int b = g >> 11;
  int sp = sidx[g];
  const float* bx = xyz + (size_t)b * NN * 3;
  float qx = bx[sp * 3 + 0], qy = bx[sp * 3 + 1], qz = bx[sp * 3 + 2];
  float qq = qx * qx + qy * qy + qz * qz;
  float bd[KNN_K]; int bi[KNN_K];
  for (int k = 0; k < KNN_K; ++k) { bd[k] = 3.4e38f; bi[k] = 0; }
  __shared__ float px[KTS * 3];
  for (int n0 = 0; n0 < NN; n0 += KTS) {
    __syncthreads();
    for (int i = threadIdx.x; i < KTS * 3; i += 256)
      px[i] = bx[n0 * 3 + i];
    __syncthreads();
    for (int i = 0; i < KTS; ++i) {
      float x = px[i * 3 + 0], y = px[i * 3 + 1], z = px[i * 3 + 2];
      float pp = x * x + y * y + z * z;
      float d  = qq - 2.0f * (qx * x + qy * y + qz * z) + pp;  // ref formula
      if (d < bd[KNN_K - 1]) {
        int p = KNN_K - 1;
        while (p > 0 && d < bd[p - 1]) { bd[p] = bd[p - 1]; bi[p] = bi[p - 1]; --p; }
        bd[p] = d; bi[p] = n0 + i;
      }
    }
  }
  for (int k = 0; k < KNN_K; ++k) knn[(size_t)g * KNN_K + k] = bi[k];
}

// ---------------- Kernel 5: centerTerm = feat_aug[sample] * W2 (WMMA) -------
__global__ void center_gemm_kernel(const float* __restrict__ feat,
                                   const float* __restrict__ xyz,
                                   const int* __restrict__ sidx,
                                   const _Float16* __restrict__ w2f,
                                   float* __restrict__ centerTerm) {
  const int t = blockIdx.x;            // 16 sampled rows per workgroup
  const int tid = threadIdx.x;
  const int w = tid >> 5, lane = tid & 31;
  __shared__ __align__(16) _Float16 As[16 * KPAD];
  __shared__ int spt[16];
  if (tid < 16) spt[tid] = sidx[t * 16 + tid];
  __syncthreads();
  for (int idx = tid; idx < 16 * KPAD; idx += 256) {
    int r = idx / KPAD, c = idx - r * KPAD;
    int grow = t * 16 + r;
    int b = grow >> 11;
    float v = 0.f;
    if (c < CAUG) {
      int n = spt[r];
      v = (c < CFEAT) ? feat[((size_t)b * NN + n) * CFEAT + c]
                      : xyz[((size_t)b * NN + n) * 3 + (c - CFEAT)];
    }
    As[idx] = (_Float16)v;
  }
  __syncthreads();
  v8f acc[2] = {};
  #pragma unroll
  for (int kt = 0; kt < NKT; ++kt) {
    v16h a  = load_afrag(As, 0, kt, lane);
    v16h b0 = load_bfrag(w2f, kt, w * 2 + 0, lane);
    v16h b1 = load_bfrag(w2f, kt, w * 2 + 1, lane);
    acc[0] = wmma_f16(a, b0, acc[0]);
    acc[1] = wmma_f16(a, b1, acc[1]);
  }
  #pragma unroll
  for (int cti = 0; cti < 2; ++cti) {
    int col = (w * 2 + cti) * 16 + (lane & 15);
    #pragma unroll
    for (int j = 0; j < 8; ++j) {
      int row = j + ((lane & 16) ? 8 : 0);
      centerTerm[(size_t)(t * 16 + row) * COUT + col] = acc[cti][j];
    }
  }
}

// ---------------- Kernel 6: zero BN stats -----------------------------------
__global__ void zero_stats_kernel(float* __restrict__ s, float* __restrict__ q) {
  int i = blockIdx.x * 256 + threadIdx.x;
  if (i < COUT) { s[i] = 0.f; q[i] = 0.f; }
}

// ------- Kernel 7/9: per-sample edge GEMM (WMMA); mode 0 = stats, 1 = out ---
__global__ void edge_gemm_kernel(const float* __restrict__ feat,
                                 const float* __restrict__ xyz,
                                 const int* __restrict__ knn,
                                 const float* __restrict__ centerTerm,
                                 const _Float16* __restrict__ wtopf,
                                 float* __restrict__ sumA,
                                 float* __restrict__ sumQ,
                                 const float* __restrict__ bnA,
                                 const float* __restrict__ bnB,
                                 float* __restrict__ out,
                                 int mode) {
  const int g = blockIdx.x;            // (b,s) group, 0 .. BB*SS-1
  const int b = g >> 11;
  const int tid = threadIdx.x;
  const int w = tid >> 5, lane = tid & 31;
  __builtin_prefetch(wtopf, 0, 3);     // global_prefetch: keep weights hot
  __shared__ __align__(16) _Float16 As[32 * KPAD];  // 20 real + 12 pad rows
  __shared__ int nbr[KNN_K];
  if (tid < KNN_K) nbr[tid] = knn[(size_t)g * KNN_K + tid];
  __syncthreads();
  for (int idx = tid; idx < 32 * KPAD; idx += 256) {
    int r = idx / KPAD, c = idx - r * KPAD;
    float v = 0.f;
    if (r < KNN_K && c < CAUG) {
      int n = nbr[r];
      v = (c < CFEAT) ? feat[((size_t)b * NN + n) * CFEAT + c]
                      : xyz[((size_t)b * NN + n) * 3 + (c - CFEAT)];
    }
    As[idx] = (_Float16)v;
  }
  __syncthreads();

  v8f acc[2][2] = {};
  #pragma unroll
  for (int kt = 0; kt < NKT; ++kt) {
    v16h a0 = load_afrag(As, 0,  kt, lane);
    v16h a1 = load_afrag(As, 16, kt, lane);
    v16h b0 = load_bfrag(wtopf, kt, w * 2 + 0, lane);
    v16h b1 = load_bfrag(wtopf, kt, w * 2 + 1, lane);
    acc[0][0] = wmma_f16(a0, b0, acc[0][0]);
    acc[0][1] = wmma_f16(a0, b1, acc[0][1]);
    acc[1][0] = wmma_f16(a1, b0, acc[1][0]);
    acc[1][1] = wmma_f16(a1, b1, acc[1][1]);
  }

  #pragma unroll
  for (int cti = 0; cti < 2; ++cti) {
    int col = (w * 2 + cti) * 16 + (lane & 15);
    float ctv = centerTerm[(size_t)g * COUT + col];
    if (mode == 0) {
      float s = 0.f, q = 0.f;
      #pragma unroll
      for (int rt = 0; rt < 2; ++rt)
        #pragma unroll
        for (int j = 0; j < 8; ++j) {
          int row = rt * 16 + j + ((lane & 16) ? 8 : 0);
          float h = acc[rt][cti][j] + ctv;
          if (row < KNN_K) { s += h; q += h * h; }
        }
      s += __shfl_xor(s, 16, 32);
      q += __shfl_xor(q, 16, 32);
      if (lane < 16) { atomicAdd(&sumA[col], s); atomicAdd(&sumQ[col], q); }
    } else {
      float a = bnA[col], bc = bnB[col];
      float m = 0.f;                       // relu output >= 0
      #pragma unroll
      for (int rt = 0; rt < 2; ++rt)
        #pragma unroll
        for (int j = 0; j < 8; ++j) {
          int row = rt * 16 + j + ((lane & 16) ? 8 : 0);
          float h = fmaxf((acc[rt][cti][j] + ctv) * a + bc, 0.f);
          if (row < KNN_K) m = fmaxf(m, h);
        }
      m = fmaxf(m, __shfl_xor(m, 16, 32));
      if (lane < 16) out[OUT_FEAT + (size_t)g * COUT + col] = m;
    }
  }
}

// ---------------- Kernel 8: finalize BN coefficients ------------------------
__global__ void finalize_stats_kernel(const float* __restrict__ s,
                                      const float* __restrict__ q,
                                      const float* __restrict__ gamma,
                                      const float* __restrict__ beta,
                                      float* __restrict__ bnA,
                                      float* __restrict__ bnB) {
  int c = threadIdx.x;
  if (c >= COUT) return;
  const float cnt = (float)(BB * SS * KNN_K);
  float mean = s[c] / cnt;
  float var  = q[c] / cnt - mean * mean;
  float a = rsqrtf(var + BN_EPS) * gamma[c];
  bnA[c] = a;
  bnB[c] = beta[c] - mean * a;
}

// ---------------- host launcher ---------------------------------------------
extern "C" void kernel_launch(void* const* d_in, const int* in_sizes, int n_in,
                              void* d_out, int out_size, void* d_ws,
                              size_t ws_size, hipStream_t stream) {
  (void)in_sizes; (void)n_in; (void)out_size; (void)ws_size;
  const float* xyz    = (const float*)d_in[0];
  const float* feat   = (const float*)d_in[1];
  const float* conv_w = (const float*)d_in[2];
  const float* gamma  = (const float*)d_in[3];
  const float* beta   = (const float*)d_in[4];
  const int*   initf  = (const int*)d_in[5];
  float* out = (float*)d_out;

  char* ws = (char*)d_ws;
  int*      sidx = (int*)(ws + OFF_SIDX);
  int*      knn  = (int*)(ws + OFF_KNN);
  _Float16* wtop = (_Float16*)(ws + OFF_WTOP);
  _Float16* w2   = (_Float16*)(ws + OFF_W2);
  float*    ct   = (float*)(ws + OFF_CT);
  float*    sum  = (float*)(ws + OFF_SUM);
  float*    sq   = (float*)(ws + OFF_SQ);
  float*    bnA  = (float*)(ws + OFF_SA);
  float*    bnB  = (float*)(ws + OFF_SB);

  fps_kernel<<<BB, 1024, 0, stream>>>(xyz, initf, sidx);
  out_misc_kernel<<<(BB * SS + 255) / 256, 256, 0, stream>>>(xyz, sidx, out);
  prep_w_kernel<<<(NKT * COUT * 32 + 255) / 256, 256, 0, stream>>>(conv_w, wtop, w2);
  knn_kernel<<<BB * SS / 256, 256, 0, stream>>>(xyz, sidx, knn);
  center_gemm_kernel<<<BB * SS / 16, 256, 0, stream>>>(feat, xyz, sidx, w2, ct);
  zero_stats_kernel<<<1, 256, 0, stream>>>(sum, sq);
  edge_gemm_kernel<<<BB * SS, 256, 0, stream>>>(feat, xyz, knn, ct, wtop,
                                                sum, sq, bnA, bnB, out, 0);
  finalize_stats_kernel<<<1, 256, 0, stream>>>(sum, sq, gamma, beta, bnA, bnB);
  edge_gemm_kernel<<<BB * SS, 256, 0, stream>>>(feat, xyz, knn, ct, wtop,
                                                sum, sq, bnA, bnB, out, 1);
}
